// GraphNN_6975026889312
// MI455X (gfx1250) — matrix operand order repaired and checked by reading, not compiled, not run on hardware
//
#include <hip/hip_runtime.h>
#include <math.h>

typedef __attribute__((ext_vector_type(16))) _Float16 v16h;
typedef __attribute__((ext_vector_type(8)))  _Float16 h8;
typedef __attribute__((ext_vector_type(8)))  float    v8f;

#define NNODES 8192
#define NEDGES 131072
#define NGRAPH 16

// ------------------------------------------------------------------
// WMMA helper (CDNA5 wave32, D = A(16x32 f16) * B(32x16 f16) + C(f32))
// ------------------------------------------------------------------
__device__ __forceinline__ v8f wmma_f16(v16h a, v16h b, v8f c) {
  return __builtin_amdgcn_wmma_f32_16x16x32_f16(false, a, false, b, (short)0, c, false, false);
}

__device__ __forceinline__ v16h cat16(h8 lo, h8 hi) {
  return __builtin_shufflevector(lo, hi, 0, 1, 2, 3, 4, 5, 6, 7, 8, 9, 10, 11, 12, 13, 14, 15);
}

// Build A fragment from 4 unconditional float4 loads (K multiple of 32).
__device__ __forceinline__ v16h frag_a_f32(const float* __restrict__ Arow, int k0, int kbA) {
  const float4* p0 = (const float4*)(Arow + k0 + kbA);
  const float4* p1 = (const float4*)(Arow + k0 + kbA + 16);
  float4 q0 = p0[0], q1 = p0[1], q2 = p1[0], q3 = p1[1];
  v16h a;
  a[0]  = (_Float16)q0.x; a[1]  = (_Float16)q0.y; a[2]  = (_Float16)q0.z; a[3]  = (_Float16)q0.w;
  a[4]  = (_Float16)q1.x; a[5]  = (_Float16)q1.y; a[6]  = (_Float16)q1.z; a[7]  = (_Float16)q1.w;
  a[8]  = (_Float16)q2.x; a[9]  = (_Float16)q2.y; a[10] = (_Float16)q2.z; a[11] = (_Float16)q2.w;
  a[12] = (_Float16)q3.x; a[13] = (_Float16)q3.y; a[14] = (_Float16)q3.z; a[15] = (_Float16)q3.w;
  return a;
}

// B fragment from the packed f16 layout: tile (kt,nt) of 32K x 16N, stored
// [col16][k32]; per lane two contiguous 16B loads, zero branches.
__device__ __forceinline__ v16h frag_b_packed(const _Float16* __restrict__ Bp,
                                              int tile, int r16, int kbB) {
  const h8* pb = (const h8*)(Bp + (size_t)tile * 512 + r16 * 32 + kbB);
  return cat16(pb[0], pb[2]);
}

// ------------------------------------------------------------------
// Kernel: pack f32 weight (K x N) into padded fragment-native f16 layout.
// ------------------------------------------------------------------
__global__ void pack_b_kernel(const float* __restrict__ B, _Float16* __restrict__ Bp,
                              int K, int N, int Kp, int Np) {
  int t = blockIdx.x * blockDim.x + threadIdx.x;
  int total = Kp * Np;
  if (t >= total) return;
  int k = t / Np, n = t % Np;
  float v = (k < K && n < N) ? B[(size_t)k * N + n] : 0.f;
  int KT = Kp >> 5;
  size_t off = ((size_t)(n >> 4) * KT + (k >> 5)) * 512 + (size_t)(n & 15) * 32 + (k & 31);
  Bp[off] = (_Float16)v;
}

// ------------------------------------------------------------------
// Kernel: build real-SH Clebsch-Gordan nonzero tables on device.
// ------------------------------------------------------------------
__device__ double dfact(int n) { double r = 1.0; for (int i = 2; i <= n; ++i) r *= (double)i; return r; }

__device__ void u_real_fill(int l, double* Ur, double* Ui) { // 7x7, stride 7
  for (int a = 0; a < 49; ++a) { Ur[a] = 0.0; Ui[a] = 0.0; }
  const double isq = 0.7071067811865476;
  Ur[l * 7 + l] = 1.0;
  for (int m = 1; m <= l; ++m) {
    double sg = (m & 1) ? -1.0 : 1.0;
    Ur[(l + m) * 7 + (l + m)] = sg * isq;
    Ur[(l + m) * 7 + (l - m)] = isq;
    Ui[(l - m) * 7 + (l - m)] = isq;
    Ui[(l - m) * 7 + (l + m)] = -sg * isq;
  }
}

__global__ void cg_init_kernel(int* __restrict__ cg_nnz, int* __restrict__ cg_idx,
                               float* __restrict__ cg_val) {
  int p = threadIdx.x;
  if (p >= 13) return;
  const int L1[13] = {0,1,2,0,1,1,2,2,0,1,1,2,2};
  const int L2[13] = {0,1,2,1,0,2,1,3,2,1,3,0,2};
  const int L3[13] = {0,0,0,1,1,1,1,1,2,2,2,2,2};
  int j1 = L1[p], j2 = L2[p], j3 = L3[p];
  int d1 = 2*j1+1, d2 = 2*j2+1, d3 = 2*j3+1;

  double Cc[5][7][5];
  for (int a = 0; a < 5; ++a) for (int b = 0; b < 7; ++b) for (int c = 0; c < 5; ++c) Cc[a][b][c] = 0.0;

  for (int m1 = -j1; m1 <= j1; ++m1)
    for (int m2 = -j2; m2 <= j2; ++m2) {
      int m3 = m1 + m2;
      if (m3 < -j3 || m3 > j3) continue;
      double pref = sqrt((double)(2*j3+1) * dfact(j1+j2-j3) * dfact(j1-j2+j3) * dfact(-j1+j2+j3) / dfact(j1+j2+j3+1));
      pref *= sqrt(dfact(j1+m1)*dfact(j1-m1)*dfact(j2+m2)*dfact(j2-m2)*dfact(j3+m3)*dfact(j3-m3));
      double s = 0.0;
      for (int k = 0; k <= j1 + j2 - j3; ++k) {
        int dens[6] = {k, j1+j2-j3-k, j1-m1-k, j2+m2-k, j3-j2+m1+k, j3-j1-m2+k};
        bool neg = false;
        for (int t = 0; t < 6; ++t) if (dens[t] < 0) neg = true;
        if (neg) continue;
        double dd = 1.0;
        for (int t = 0; t < 6; ++t) dd *= dfact(dens[t]);
        s += ((k & 1) ? -1.0 : 1.0) / dd;
      }
      Cc[m1+j1][m2+j2][m3+j3] = pref * s;
    }

  double U1r[49], U1i[49], U2r[49], U2i[49], U3r[49], U3i[49];
  u_real_fill(j1, U1r, U1i);
  u_real_fill(j2, U2r, U2i);
  u_real_fill(j3, U3r, U3i);

  double Cr[5][7][5], Ci[5][7][5];
  double nr = 0.0, ni = 0.0;
  for (int i = 0; i < d1; ++i)
    for (int j = 0; j < d2; ++j)
      for (int k = 0; k < d3; ++k) {
        double ar = 0.0, ai = 0.0;
        for (int m = 0; m < d1; ++m)
          for (int n = 0; n < d2; ++n)
            for (int o = 0; o < d3; ++o) {
              double cv = Cc[m][n][o];
              if (cv == 0.0) continue;
              double t1r = U1r[i*7+m]*U2r[j*7+n] - U1i[i*7+m]*U2i[j*7+n];
              double t1i = U1r[i*7+m]*U2i[j*7+n] + U1i[i*7+m]*U2r[j*7+n];
              double t2r = t1r*U3r[k*7+o] + t1i*U3i[k*7+o];
              double t2i = t1i*U3r[k*7+o] - t1r*U3i[k*7+o];
              ar += t2r * cv;
              ai += t2i * cv;
            }
        Cr[i][j][k] = ar; Ci[i][j][k] = ai;
        nr += ar*ar; ni += ai*ai;
      }
  bool useR = (nr >= ni);
  double nsel = sqrt(useR ? nr : ni);
  double scl = sqrt((double)d3) / nsel;
  int cnt = 0;
  for (int i = 0; i < d1; ++i)
    for (int j = 0; j < d2; ++j)
      for (int k = 0; k < d3; ++k) {
        double v = (useR ? Cr[i][j][k] : Ci[i][j][k]) * scl;
        if (fabs(v) > 1e-10) {
          cg_idx[p*128 + cnt] = i | (j << 4) | (k << 8);
          cg_val[p*128 + cnt] = (float)v;
          ++cnt;
        }
      }
  cg_nnz[p] = cnt;
}

// ------------------------------------------------------------------
// Kernel: per-edge geometry: spherical harmonics (l<=3) + radial basis
// ------------------------------------------------------------------
__global__ void edge_geom_kernel(const float* __restrict__ Rij, float* __restrict__ sh,
                                 float* __restrict__ rb, int E) {
  int e = blockIdx.x * blockDim.x + threadIdx.x;
  if (e >= E) return;
  float x = Rij[e*3+0] * 0.2f, y = Rij[e*3+1] * 0.2f, z = Rij[e*3+2] * 0.2f; // /CUTOFF
  float len = sqrtf(x*x + y*y + z*z);
  float safe = (len > 0.f) ? len : 1.f;
  float ux = x / safe, uy = y / safe, uz = z / safe;
  float* S = sh + (size_t)e * 16;
  const float s3 = 1.7320508075688772f, s15 = 3.872983346207417f, s5 = 2.23606797749979f;
  const float ca = 2.091650066335189f, cb = 10.246950765959598f, cc = 1.6201851746019651f, cd = 2.6457513110645907f;
  S[0] = 1.f;
  S[1] = s3*uy; S[2] = s3*uz; S[3] = s3*ux;
  S[4] = s15*ux*uy; S[5] = s15*uy*uz; S[6] = 0.5f*s5*(3.f*uz*uz - 1.f);
  S[7] = s15*ux*uz; S[8] = 0.5f*s15*(ux*ux - uy*uy);
  S[9]  = ca*uy*(3.f*ux*ux - uy*uy);
  S[10] = cb*ux*uy*uz;
  S[11] = cc*uy*(5.f*uz*uz - 1.f);
  S[12] = 0.5f*cd*(5.f*uz*uz*uz - 3.f*uz);
  S[13] = cc*ux*(5.f*uz*uz - 1.f);
  S[14] = 0.5f*cb*uz*(ux*ux - uy*uy);
  S[15] = ca*ux*(ux*ux - 3.f*uy*uy);
  float env = (len < 1.f) ? (1.f - 6.f*len*len + 8.f*len*len*len - 3.f*len*len*len*len) : 0.f;
  const float sq2 = 1.4142135623730951f, pi = 3.14159265358979323846f;
  for (int k = 1; k <= 8; ++k) {
    float v = (len == 0.f) ? 0.f : sq2 * __sinf(pi * (float)k * len) / safe * env;
    rb[(size_t)e*8 + (k-1)] = v;
  }
}

// ------------------------------------------------------------------
// Kernel: WMMA GEMM  C[M,N] = scale * A[M,K](f32) @ Bp(packed f16)
// K multiple of 32, N multiple of 16. One wave per 16x16 tile.
// Optional row gather on A (row -> ridx[row]).
// ------------------------------------------------------------------
__global__ __launch_bounds__(32)
void wmma_gemm_kernel(const float* __restrict__ A, const int* __restrict__ ridx,
                      const _Float16* __restrict__ Bp, float* __restrict__ C,
                      int N, int K, float scale) {
  int lane = threadIdx.x & 31;
  int nt = blockIdx.x;
  int m0 = blockIdx.y * 16;
  int r16 = lane & 15;
  int row = m0 + r16;
  int arow = ridx ? ridx[row] : row;
  const float* Arow = A + (size_t)arow * K;
  int kbA = (lane >> 4) << 3;   // 0 or 8 (A interleave per ISA table)
  int kbB = (lane >> 4) << 3;   // 0 or 8 (B interleave per ISA table)
  int KT = K >> 5;
  v8f acc = {};
  for (int kt = 0; kt < KT; ++kt) {
    v16h a = frag_a_f32(Arow, kt << 5, kbA);
    v16h b = frag_b_packed(Bp, nt * KT + kt, r16, kbB);
    acc = wmma_f16(a, b, acc);
  }
  int rbase = (lane >> 4) << 3;
  int col = nt * 16 + r16;
  #pragma unroll
  for (int r = 0; r < 8; ++r) {
    int m = m0 + rbase + r;
    C[(size_t)m * N + col] = acc[r] * scale;
  }
}

// ------------------------------------------------------------------
// Kernel: fused radial MLP on edges via WMMA, 1 wave = 16 edges.
// rb(E,8) -> silu(@W0/s8) -> silu(@W1/s64) -> (@W2/s64) -> mix(E,200) f16
// All weights pre-packed (zero-padded) f16; all loads unconditional.
// ------------------------------------------------------------------
__global__ __launch_bounds__(32)
void radial_mlp_kernel(const float* __restrict__ rb, const _Float16* __restrict__ W0p,
                       const _Float16* __restrict__ W1p, const _Float16* __restrict__ W2p,
                       _Float16* __restrict__ mix, int E) {
  __shared__ __attribute__((aligned(16))) _Float16 s_in[16][32];
  __shared__ __attribute__((aligned(16))) _Float16 s_h1[16][64];
  __shared__ __attribute__((aligned(16))) _Float16 s_h2[16][64];
  int lane = threadIdx.x & 31;
  int e0 = blockIdx.x * 16;
  for (int idx = lane; idx < 16 * 32; idx += 32) {
    int rr = idx >> 5, cc = idx & 31;
    float v = (cc < 8) ? rb[(size_t)(e0 + rr) * 8 + cc] : 0.f;
    s_in[rr][cc] = (_Float16)v;
  }
  __syncthreads();
  const float s8 = 0.35355339059327373f;   // 1/sqrt(8)
  const float s64 = 0.125f;                // 1/sqrt(64)
  int r16 = lane & 15;
  int kbA = (lane >> 4) << 3;
  int kbB = (lane >> 4) << 3;
  int rbase = kbA;
  // ---- stage 1: (16x8 pad 32) @ (32x64 packed, KT=1) ----
  {
    h8 lo = *(const h8*)&s_in[r16][kbA];
    h8 hi = *(const h8*)&s_in[r16][kbA + 16];
    v16h a = cat16(lo, hi);
    for (int nt = 0; nt < 4; ++nt) {
      v16h b = frag_b_packed(W0p, nt, r16, kbB);
      v8f c = {};
      c = wmma_f16(a, b, c);
      int col = nt * 16 + r16;
      #pragma unroll
      for (int r = 0; r < 8; ++r) {
        float y = c[r] * s8;
        y = y / (1.f + __expf(-y));            // silu
        s_h1[rbase + r][col] = (_Float16)y;
      }
    }
  }
  __syncthreads();
  // ---- stage 2: (16x64) @ (64x64 packed, KT=2) ----
  for (int nt = 0; nt < 4; ++nt) {
    v8f c = {};
    #pragma unroll
    for (int ks = 0; ks < 2; ++ks) {
      int k0 = ks * 32;
      h8 lo = *(const h8*)&s_h1[r16][k0 + kbA];
      h8 hi = *(const h8*)&s_h1[r16][k0 + kbA + 16];
      v16h a = cat16(lo, hi);
      v16h b = frag_b_packed(W1p, nt * 2 + ks, r16, kbB);
      c = wmma_f16(a, b, c);
    }
    int col = nt * 16 + r16;
    #pragma unroll
    for (int r = 0; r < 8; ++r) {
      float y = c[r] * s64;
      y = y / (1.f + __expf(-y));
      s_h2[rbase + r][col] = (_Float16)y;
    }
  }
  __syncthreads();
  // ---- stage 3: (16x64) @ (64x200 pad 208 packed, KT=2, NT=13) ----
  for (int nt = 0; nt < 13; ++nt) {
    v8f c = {};
    #pragma unroll
    for (int ks = 0; ks < 2; ++ks) {
      int k0 = ks * 32;
      h8 lo = *(const h8*)&s_h2[r16][k0 + kbA];
      h8 hi = *(const h8*)&s_h2[r16][k0 + kbA + 16];
      v16h a = cat16(lo, hi);
      v16h b = frag_b_packed(W2p, nt * 2 + ks, r16, kbB);
      c = wmma_f16(a, b, c);
    }
    int col = nt * 16 + r16;
    if (col < 200) {
      #pragma unroll
      for (int r = 0; r < 8; ++r)
        mix[(size_t)(e0 + rbase + r) * 200 + col] = (_Float16)(c[r] * s64);
    }
  }
}

// ------------------------------------------------------------------
// Kernel: per-node up-projection h[l] and species-dependent skip[l].
// Feature layout per node: [l0:32x1][l1:8x3][l2:4x5] = 76 floats.
// ------------------------------------------------------------------
__global__ void node_linear_kernel(const float* __restrict__ feats, const float* __restrict__ embed,
                                   const int* __restrict__ species,
                                   const float* __restrict__ up0, const float* __restrict__ up1,
                                   const float* __restrict__ up2,
                                   const float* __restrict__ sk0, const float* __restrict__ sk1,
                                   const float* __restrict__ sk2,
                                   float* __restrict__ h, float* __restrict__ skip, int layer0) {
  int t = blockIdx.x * blockDim.x + threadIdx.x;
  int n = t / 44;
  if (n >= NNODES) return;
  int lo = t % 44;
  int l, o;
  if (lo < 32) { l = 0; o = lo; }
  else if (lo < 40) { l = 1; o = lo - 32; }
  else { l = 2; o = lo - 40; }
  const int hoff[3] = {0, 32, 56};
  int d = 2 * l + 1;
  int sp = species[n];
  if (layer0) {
    if (l != 0) {
      for (int i = 0; i < d; ++i) {
        h[(size_t)n*76 + hoff[l] + o*d + i] = 0.f;
        skip[(size_t)n*76 + hoff[l] + o*d + i] = 0.f;
      }
      return;
    }
    const float* x = embed + (size_t)sp * 128;
    const float* ws = sk0 + (size_t)sp * 128 * 32;
    float ah = 0.f, as = 0.f;
    for (int m = 0; m < 128; ++m) {
      float xv = x[m];
      ah += xv * up0[m*32 + o];
      as += xv * ws[m*32 + o];
    }
    const float s = 0.08838834764831845f; // 1/sqrt(128)
    h[(size_t)n*76 + o] = ah * s;
    skip[(size_t)n*76 + o] = as * s;
  } else {
    const int mis[3] = {32, 8, 4};
    int mi = mis[l];
    const float* wu = (l == 0) ? up0 : ((l == 1) ? up1 : up2);
    const float* ws = ((l == 0) ? sk0 : ((l == 1) ? sk1 : sk2)) + (size_t)sp * mi * mi;
    float s = rsqrtf((float)mi);
    for (int i = 0; i < d; ++i) {
      float ah = 0.f, as = 0.f;
      for (int m = 0; m < mi; ++m) {
        float xv = feats[(size_t)n*76 + hoff[l] + m*d + i];
        ah += xv * wu[m*mi + o];
        as += xv * ws[m*mi + o];
      }
      h[(size_t)n*76 + hoff[l] + o*d + i] = ah * s;
      skip[(size_t)n*76 + hoff[l] + o*d + i] = as * s;
    }
  }
}

// ------------------------------------------------------------------
// Kernel: per-edge tensor product + gate + scatter-add (f32 atomics).
// agg layout per node: [l0:76x1][l1:64x3][l2:60x5] = 568 floats.
// ------------------------------------------------------------------
__global__ void tp_scatter_kernel(const float* __restrict__ h, const float* __restrict__ sh,
                                  const _Float16* __restrict__ mix,
                                  const int* __restrict__ senders, const int* __restrict__ receivers,
                                  const int* __restrict__ cg_nnz, const int* __restrict__ cg_idx,
                                  const float* __restrict__ cg_val,
                                  float* __restrict__ agg, int E) {
  int e = blockIdx.x * blockDim.x + threadIdx.x;
  if (e >= E) return;
  int sN = senders[e], rN = receivers[e];
  __builtin_prefetch(h + (size_t)sN * 76, 0, 3);   // global_prefetch on gathered row
  float hb[76];
  {
    const float4* h4 = (const float4*)(h + (size_t)sN * 76);   // 76*4B row, 16B aligned
    float4* d4 = (float4*)hb;
    #pragma unroll
    for (int i = 0; i < 19; ++i) d4[i] = h4[i];
  }
  float sb[16];
  {
    const float4* s4 = (const float4*)(sh + (size_t)e * 16);
    float4* d4 = (float4*)sb;
    #pragma unroll
    for (int i = 0; i < 4; ++i) d4[i] = s4[i];
  }
  const _Float16* g = mix + (size_t)e * 200;
  float* ar = agg + (size_t)rN * 568;
  // identity blocks
  for (int m = 0; m < 32; ++m) atomicAdd(ar + m, hb[m] * (float)g[m]);
  for (int m = 0; m < 8; ++m) {
    float gg = (float)g[76 + m];
    for (int k = 0; k < 3; ++k) atomicAdd(ar + 76 + m*3 + k, hb[32 + m*3 + k] * gg);
  }
  for (int m = 0; m < 4; ++m) {
    float gg = (float)g[140 + m];
    for (int k = 0; k < 5; ++k) atomicAdd(ar + 268 + m*5 + k, hb[56 + m*5 + k] * gg);
  }
  // CG paths
  const int PL1[13]  = {0,1,2,0,1,1,2,2,0,1,1,2,2};
  const int PL2[13]  = {0,1,2,1,0,2,1,3,2,1,3,0,2};
  const int PL3[13]  = {0,0,0,1,1,1,1,1,2,2,2,2,2};
  const int PROW[13] = {32,64,72,8,40,48,56,60,4,36,44,52,56};
  const int HOFF[3]  = {0, 32, 56};
  const int SHOFF[4] = {0, 1, 4, 9};
  const int GOFF[3]  = {0, 76, 140};
  const int ABASE[3] = {0, 76, 268};
  const int MUL[3]   = {32, 8, 4};
  for (int pth = 0; pth < 13; ++pth) {
    int l1 = PL1[pth], l2 = PL2[pth], l3 = PL3[pth];
    int d1 = 2*l1 + 1, d3 = 2*l3 + 1;
    int mul = MUL[l1];
    int nnz = cg_nnz[pth];
    const int* pi = cg_idx + pth * 128;
    const float* pv = cg_val + pth * 128;
    const float* hrow = hb + HOFF[l1];
    const float* srow = sb + SHOFF[l2];
    int rowbase = PROW[pth];
    for (int m = 0; m < mul; ++m) {
      float out[5] = {0.f, 0.f, 0.f, 0.f, 0.f};
      for (int t = 0; t < nnz; ++t) {
        int pk = pi[t];
        out[(pk >> 8) & 15] += pv[t] * hrow[m*d1 + (pk & 15)] * srow[(pk >> 4) & 15];
      }
      float gg = (float)g[GOFF[l3] + rowbase + m];
      float* dst = ar + ABASE[l3] + (size_t)(rowbase + m) * d3;
      for (int k = 0; k < d3; ++k) atomicAdd(dst + k, gg * out[k]);
    }
  }
}

// ------------------------------------------------------------------
// Kernel: down-projection h2[l] = scale * agg[l] @ down_l
// ------------------------------------------------------------------
__global__ void down_kernel(const float* __restrict__ agg, const float* __restrict__ W,
                            float* __restrict__ h2, int mt, int hid, int d, int abase,
                            int hoff, float scale) {
  int t = blockIdx.x * blockDim.x + threadIdx.x;
  int n = t / hid;
  if (n >= NNODES) return;
  int o = t % hid;
  for (int i = 0; i < d; ++i) {
    float acc = 0.f;
    for (int m = 0; m < mt; ++m)
      acc += agg[(size_t)n * 568 + abase + m*d + i] * W[m*hid + o];
    h2[(size_t)n * 76 + hoff + o*d + i] = acc * scale;
  }
}

// ------------------------------------------------------------------
// Kernel: feats[l] = scale * einsum(h2[n,m,i], Wn[n,m,o]) + skip[l]
// ------------------------------------------------------------------
__global__ void feat_kernel(const float* __restrict__ h2, const float* __restrict__ Wn,
                            const float* __restrict__ skip, float* __restrict__ feats,
                            int hid, int d, int hoff, float scale) {
  int t = blockIdx.x * blockDim.x + threadIdx.x;
  int n = t / hid;
  if (n >= NNODES) return;
  int o = t % hid;
  const float* wn = Wn + (size_t)n * hid * hid;
  for (int i = 0; i < d; ++i) {
    float acc = 0.f;
    for (int m = 0; m < hid; ++m)
      acc += h2[(size_t)n * 76 + hoff + m*d + i] * wn[m*hid + o];
    feats[(size_t)n * 76 + hoff + o*d + i] =
        acc * scale + skip[(size_t)n * 76 + hoff + o*d + i];
  }
}

// ------------------------------------------------------------------
// Kernel: per-layer readout, accumulates node energy.
// ------------------------------------------------------------------
__global__ void readout_kernel(const float* __restrict__ feats, const float* __restrict__ qfeat,
                               const float* __restrict__ wq, const float* __restrict__ head_w,
                               const float* __restrict__ head_b, float* __restrict__ node_energy) {
  int n = blockIdx.x * blockDim.x + threadIdx.x;
  if (n >= NNODES) return;
  float f0[32];
  for (int o = 0; o < 32; ++o) f0[o] = feats[(size_t)n * 76 + o];
  const float s32 = 0.17677669529663687f; // 1/sqrt(32)
  const float s8  = 0.35355339059327373f; // 1/sqrt(8)
  float e = 0.f;
  for (int q = 0; q < 8; ++q) {
    float acc = 0.f;
    for (int o = 0; o < 32; ++o) acc += f0[o] * qfeat[o*8 + q];
    float cp = __cosf(wq[q]) * __cosf(wq[8 + q]) * __cosf(wq[16 + q]);
    e += __cosf(acc * s32) * cp * head_w[q];
  }
  node_energy[n] += e * s8 + head_b[0];
}

// ------------------------------------------------------------------
// Kernel: graph reduction (contiguous 512 nodes / graph) + variance.
// ------------------------------------------------------------------
__global__ void reduce_kernel(const float* __restrict__ ne, float* __restrict__ out) {
  __shared__ float sdata[256];
  int g = blockIdx.x, t = threadIdx.x;
  float acc = 0.f;
  for (int i = t; i < 512; i += 256) acc += ne[g * 512 + i];
  sdata[t] = acc;
  __syncthreads();
  for (int s = 128; s > 0; s >>= 1) {
    if (t < s) sdata[t] += sdata[t + s];
    __syncthreads();
  }
  if (t == 0) { out[g] = sdata[0]; out[NGRAPH + g] = 1.0f; }
}

// ------------------------------------------------------------------
// Host launcher
// ------------------------------------------------------------------
struct LayerP {
  const float *down0, *down1, *down2, *head_b, *head_w, *mlp0, *mlp1, *mlp2, *qfeat;
  const float *skip0, *skip1, *skip2, *up0, *up1, *up2, *wq, *xlin0, *xlin1, *xlin2;
};

extern "C" void kernel_launch(void* const* d_in, const int* in_sizes, int n_in,
                              void* d_out, int out_size, void* d_ws, size_t ws_size,
                              hipStream_t stream) {
  // ---- input map (jax pytree: dict keys sorted; params after Rij) ----
  const float* Rij = (const float*)d_in[0];
  const int LBASE[4] = {1, 16, 35, 54}; // layer0 has 15 leaves, layers 1-3 have 19
  LayerP LP[4];
  for (int i = 0; i < 4; ++i) {
    int b = LBASE[i];
    LP[i].down0 = (const float*)d_in[b + 0];
    LP[i].down1 = (const float*)d_in[b + 1];
    LP[i].down2 = (const float*)d_in[b + 2];
    LP[i].head_b = (const float*)d_in[b + 3];
    LP[i].head_w = (const float*)d_in[b + 4];
    LP[i].mlp0 = (const float*)d_in[b + 5];
    LP[i].mlp1 = (const float*)d_in[b + 6];
    LP[i].mlp2 = (const float*)d_in[b + 7];
    LP[i].qfeat = (const float*)d_in[b + 8];
    LP[i].skip0 = (const float*)d_in[b + 9];
    if (i == 0) {
      LP[i].skip1 = nullptr; LP[i].skip2 = nullptr;
      LP[i].up0 = (const float*)d_in[b + 10]; LP[i].up1 = nullptr; LP[i].up2 = nullptr;
      LP[i].wq = (const float*)d_in[b + 11];
      LP[i].xlin0 = (const float*)d_in[b + 12];
      LP[i].xlin1 = (const float*)d_in[b + 13];
      LP[i].xlin2 = (const float*)d_in[b + 14];
    } else {
      LP[i].skip1 = (const float*)d_in[b + 10];
      LP[i].skip2 = (const float*)d_in[b + 11];
      LP[i].up0 = (const float*)d_in[b + 12];
      LP[i].up1 = (const float*)d_in[b + 13];
      LP[i].up2 = (const float*)d_in[b + 14];
      LP[i].wq = (const float*)d_in[b + 15];
      LP[i].xlin0 = (const float*)d_in[b + 16];
      LP[i].xlin1 = (const float*)d_in[b + 17];
      LP[i].xlin2 = (const float*)d_in[b + 18];
    }
  }
  const float* embed = (const float*)d_in[73];
  const float* x_lin = (const float*)d_in[74];
  const int* senders = (const int*)d_in[75];
  const int* receivers = (const int*)d_in[76];
  const int* species = (const int*)d_in[77];
  // d_in[78] node_graph (contiguous), d_in[79] n_node (all 512) -> implicit

  // ---- workspace partition ----
  size_t off = 0;
  auto carve = [&](size_t bytes) -> char* {
    char* p = (char*)d_ws + off;
    off = (off + bytes + 255) & ~(size_t)255;
    return p;
  };
  float* sh       = (float*)carve((size_t)NEDGES * 16 * 4);   // 8 MB
  float* rb       = (float*)carve((size_t)NEDGES * 8 * 4);    // 4 MB
  _Float16* mix   = (_Float16*)carve((size_t)NEDGES * 200 * 2); // 52 MB (f16 -> L2 resident)
  float* x0       = (float*)carve((size_t)NNODES * 128 * 4);
  float* feats    = (float*)carve((size_t)NNODES * 76 * 4);
  float* hbuf     = (float*)carve((size_t)NNODES * 76 * 4);
  float* skip     = (float*)carve((size_t)NNODES * 76 * 4);
  float* agg      = (float*)carve((size_t)NNODES * 568 * 4);  // 18.6 MB
  float* h2       = (float*)carve((size_t)NNODES * 76 * 4);
  float* Wn       = (float*)carve((size_t)NNODES * 1024 * 4); // 33.6 MB (reused per l)
  float* node_e   = (float*)carve((size_t)NNODES * 4);
  int*   cg_nnz   = (int*)carve(13 * 4);
  int*   cg_idx   = (int*)carve(13 * 128 * 4);
  float* cg_val   = (float*)carve(13 * 128 * 4);
  // packed f16 weights
  _Float16* xlp = (_Float16*)carve((size_t)128 * 128 * 2);
  _Float16 *mlp0p[4], *mlp1p[4], *mlp2p[4], *xlinp[4][3];
  for (int i = 0; i < 4; ++i) {
    mlp0p[i] = (_Float16*)carve((size_t)32 * 64 * 2);
    mlp1p[i] = (_Float16*)carve((size_t)64 * 64 * 2);
    mlp2p[i] = (_Float16*)carve((size_t)64 * 208 * 2);
    xlinp[i][0] = (_Float16*)carve((size_t)128 * 1024 * 2);
    xlinp[i][1] = (_Float16*)carve((size_t)128 * 64 * 2);
    xlinp[i][2] = (_Float16*)carve((size_t)128 * 16 * 2);
  }
  (void)ws_size; (void)in_sizes; (void)n_in; (void)out_size;

  const int HID[3] = {32, 8, 4};
  const int MT[3] = {76, 64, 60};
  const int DIM[3] = {1, 3, 5};
  const int ABASE[3] = {0, 76, 268};
  const int HOFF[3] = {0, 32, 56};
  const float feat_scale[3] = {0.015625f, 0.03125f, 0.04419417382415922f}; // 1/sqrt(128*HID)

  auto pack = [&](const float* B, _Float16* Bp, int K, int N, int Kp, int Np) {
    int total = Kp * Np;
    pack_b_kernel<<<(total + 255) / 256, 256, 0, stream>>>(B, Bp, K, N, Kp, Np);
  };

  hipMemsetAsync(node_e, 0, (size_t)NNODES * 4, stream);
  cg_init_kernel<<<1, 16, 0, stream>>>(cg_nnz, cg_idx, cg_val);
  edge_geom_kernel<<<NEDGES / 256, 256, 0, stream>>>(Rij, sh, rb, NEDGES);

  // pack all weights once per call
  pack(x_lin, xlp, 128, 128, 128, 128);
  for (int i = 0; i < 4; ++i) {
    pack(LP[i].mlp0, mlp0p[i], 8, 64, 32, 64);
    pack(LP[i].mlp1, mlp1p[i], 64, 64, 64, 64);
    pack(LP[i].mlp2, mlp2p[i], 64, 200, 64, 208);
    pack(LP[i].xlin0, xlinp[i][0], 128, 1024, 128, 1024);
    pack(LP[i].xlin1, xlinp[i][1], 128, 64, 128, 64);
    pack(LP[i].xlin2, xlinp[i][2], 128, 16, 128, 16);
  }

  // x0 = embed[species] @ x_lin / sqrt(128)  -- WMMA GEMM with row gather
  {
    dim3 grid(128 / 16, NNODES / 16);
    wmma_gemm_kernel<<<grid, 32, 0, stream>>>(embed, species, xlp, x0,
                                              128, 128, 0.08838834764831845f);
  }

  for (int layer = 0; layer < 4; ++layer) {
    const LayerP& P = LP[layer];
    // up-projection + skip
    node_linear_kernel<<<(NNODES * 44 + 255) / 256, 256, 0, stream>>>(
        feats, embed, species, P.up0, P.up1, P.up2, P.skip0, P.skip1, P.skip2,
        hbuf, skip, layer == 0 ? 1 : 0);
    // radial MLP (WMMA, packed weights)
    radial_mlp_kernel<<<NEDGES / 16, 32, 0, stream>>>(rb, mlp0p[layer], mlp1p[layer],
                                                      mlp2p[layer], mix, NEDGES);
    // tensor-product message passing
    hipMemsetAsync(agg, 0, (size_t)NNODES * 568 * 4, stream);
    tp_scatter_kernel<<<NEDGES / 256, 256, 0, stream>>>(hbuf, sh, mix, senders, receivers,
                                                        cg_nnz, cg_idx, cg_val, agg, NEDGES);
    // down-projections (fold both 1/sqrt(16) factors: 1/16)
    const float* dwn[3] = {P.down0, P.down1, P.down2};
    for (int l = 0; l < 3; ++l) {
      float ds = 1.0f / (16.0f * sqrtf((float)MT[l]));
      down_kernel<<<(NNODES * HID[l] + 255) / 256, 256, 0, stream>>>(
          agg, dwn[l], h2, MT[l], HID[l], DIM[l], ABASE[l], HOFF[l], ds);
    }
    // xlin: Wn = x0 @ xlin_l  (WMMA GEMM), then feats = contract + skip
    for (int l = 0; l < 3; ++l) {
      int nn = HID[l] * HID[l];
      dim3 grid(nn / 16, NNODES / 16);
      wmma_gemm_kernel<<<grid, 32, 0, stream>>>(x0, nullptr, xlinp[layer][l], Wn,
                                                nn, 128, 1.0f);
      feat_kernel<<<(NNODES * HID[l] + 255) / 256, 256, 0, stream>>>(
          h2, Wn, skip, feats, HID[l], DIM[l], HOFF[l], feat_scale[l]);
    }
    // readout
    readout_kernel<<<NNODES / 256, 256, 0, stream>>>(feats, P.qfeat, P.wq, P.head_w,
                                                     P.head_b, node_e);
  }
  reduce_kernel<<<NGRAPH, 256, 0, stream>>>(node_e, (float*)d_out);
}
